// Generator_35313221107777
// MI455X (gfx1250) — compile-verified, matrix operand order checked by reading, not run on hardware
//
#include <hip/hip_runtime.h>
#include <math.h>

// ---------------------------------------------------------------------------
// Model dims (from reference)
// ---------------------------------------------------------------------------
#define BS   16
#define SEQ  64
#define EMB  256
#define HID  1024
#define VOC  30000
#define NH   8
#define DH   32

typedef __attribute__((ext_vector_type(16))) _Float16 v16h;
typedef __attribute__((ext_vector_type(8)))  _Float16 v8h;
typedef __attribute__((ext_vector_type(8)))  float    v8f;

// ---------------------------------------------------------------------------
// f32 -> f16 conversion (weights once per launch; noise once)
// ---------------------------------------------------------------------------
__global__ __launch_bounds__(256) void cvt_f16_kernel(const float* s, _Float16* d, int n) {
  int i = blockIdx.x * 256 + threadIdx.x;
  if (i < n) d[i] = (_Float16)s[i];
}

// ---------------------------------------------------------------------------
// GEMM: C[M,N] = act(A[M,K] @ W[N,K]^T + bias), A/W in f16, f32 accumulate.
// One wave per 16x16 tile; V_WMMA_F32_16X16X32_F16.
// Fragment layout per ISA: halves 0..7 -> K = grp*8+0..7,
//                          halves 8..15 -> K = 16+grp*8+0..7  (grp = lane/16)
// Row mapping rowBase = (tm/sTiles)*SEQ + (tm%sTiles)*16 supports causal
// truncation (sTiles = t/16+1 per step) and degenerates for full/head GEMMs.
// C (f32) and Ch (f16 mirror) both optional.
// ---------------------------------------------------------------------------
template <int ACT>
__global__ __launch_bounds__(256) void gemm16_kernel(
    const _Float16* A, int lda, const _Float16* W, const float* bias,
    float* C, _Float16* Ch, int ldc, int mTiles, int sTiles, int N, int K)
{
  const int wave = threadIdx.x >> 5;
  const int lane = threadIdx.x & 31;
  const int tilesN = N >> 4;
  const int tile = blockIdx.x * 8 + wave;
  if (tile >= mTiles * tilesN) return;          // wave-uniform: EXEC stays full
  const int tm = tile / tilesN;
  const int tn = tile - tm * tilesN;
  const int rowBase = (tm / sTiles) * SEQ + (tm % sTiles) * 16;
  const int grp = lane >> 4;
  const int lm  = lane & 15;

  const _Float16* arow = A + (size_t)(rowBase + lm) * (size_t)lda;
  const _Float16* wrow = W + (size_t)(tn * 16 + lm) * (size_t)K;

  v8f acc = {};
  int k0 = 0;
  for (; k0 + 32 <= K; k0 += 32) {
    const _Float16* ap = arow + k0 + (grp << 3);
    const _Float16* wp = wrow + k0 + (grp << 3);
    __builtin_prefetch(ap + 64, 0, 0);          // global_prefetch_b8
    __builtin_prefetch(wp + 64, 0, 0);
    v8h a0 = *(const v8h*)(ap);                 // 16B: K = grp*8 .. +7
    v8h a1 = *(const v8h*)(ap + 16);            // 16B: K = 16+grp*8 .. +7
    v8h b0 = *(const v8h*)(wp);
    v8h b1 = *(const v8h*)(wp + 16);
    v16h af = __builtin_shufflevector(a0, a1, 0,1,2,3,4,5,6,7,8,9,10,11,12,13,14,15);
    v16h bf = __builtin_shufflevector(b0, b1, 0,1,2,3,4,5,6,7,8,9,10,11,12,13,14,15);
    acc = __builtin_amdgcn_wmma_f32_16x16x32_f16(
        false, af, false, bf, (short)0, acc, false, false);
  }
  if (k0 < K) {                                 // K tail (30000 % 32 == 16)
    v16h af = {}, bf = {};
#pragma unroll
    for (int i = 0; i < 16; ++i) {
      int kk = ((i >> 3) << 4) + (grp << 3) + (i & 7) + k0;
      if (kk < K) { af[i] = arow[kk]; bf[i] = wrow[kk]; }
    }
    acc = __builtin_amdgcn_wmma_f32_16x16x32_f16(
        false, af, false, bf, (short)0, acc, false, false);
  }

  const int col = tn * 16 + lm;
  const float bv = bias ? bias[col] : 0.0f;
#pragma unroll
  for (int j = 0; j < 8; ++j) {                 // acc[j] -> row j + 8*grp
    int row = rowBase + j + (grp << 3);
    float v = acc[j] + bv;
    if (ACT) v = fmaxf(v, 0.0f);
    size_t off = (size_t)row * (size_t)ldc + col;
    if (C)  C[off]  = v;
    if (Ch) Ch[off] = (_Float16)v;
  }
}

static void gemm(const _Float16* A, int lda, const _Float16* W, const float* bias,
                 float* C, _Float16* Ch, int ldc, int mTiles, int sTiles,
                 int N, int K, int act, hipStream_t st)
{
  int tiles  = mTiles * (N >> 4);
  int blocks = (tiles + 7) / 8;
  if (act) gemm16_kernel<1><<<blocks, 256, 0, st>>>(A, lda, W, bias, C, Ch, ldc, mTiles, sTiles, N, K);
  else     gemm16_kernel<0><<<blocks, 256, 0, st>>>(A, lda, W, bias, C, Ch, ldc, mTiles, sTiles, N, K);
}

// ---------------------------------------------------------------------------
// Positional encoding (once per launch)
// ---------------------------------------------------------------------------
__global__ void pe_init_kernel(float* pe) {
  int s = blockIdx.x, e = threadIdx.x;
  int pair = e >> 1;
  float f   = __expf((float)(2 * pair) * (-9.210340371976184f / 256.0f)); // -ln(1e4)/E
  float ang = (float)s * f;
  pe[s * EMB + e] = (e & 1) ? cosf(ang) : sinf(ang);
}

__global__ void zero_kernel(float* p, int n) {
  int i = blockIdx.x * 256 + threadIdx.x;
  if (i < n) p[i] = 0.0f;
}

// Y[:,t] = seeds[:,t] ; x = Y + pe (f32 + f16 mirror)
__global__ void step_begin_kernel(float* Y, const float* seeds, const float* pe,
                                  float* x, _Float16* xh, int t) {
  int i = blockIdx.x * 256 + threadIdx.x;       // over BS*SEQ*EMB
  int e = i & (EMB - 1);
  int s = (i >> 8) & (SEQ - 1);
  float y;
  if (s == t) { y = seeds[i]; Y[i] = y; } else { y = Y[i]; }
  float v = y + pe[(s << 8) + e];
  x[i]  = v;
  xh[i] = (_Float16)v;
}

// ---------------------------------------------------------------------------
// x = LN(pr + x) * g + b over rows s <= t; writes f32 + f16 mirror.
// One wave per 256-wide row.
// ---------------------------------------------------------------------------
__global__ __launch_bounds__(256) void add_ln_kernel(
    const float* pr, const float* res, const float* g, const float* b,
    float* out, _Float16* outh, int sRows)
{
  int widx = blockIdx.x * 8 + (threadIdx.x >> 5);
  if (widx >= BS * sRows) return;
  int lane = threadIdx.x & 31;
  int row  = (widx / sRows) * SEQ + (widx % sRows);
  const float* xr = pr  + (size_t)row * EMB;
  const float* rr = res + (size_t)row * EMB;
  float v[8];
  float s = 0.0f;
#pragma unroll
  for (int i = 0; i < 8; ++i) { v[i] = xr[lane * 8 + i] + rr[lane * 8 + i]; s += v[i]; }
#pragma unroll
  for (int off = 16; off > 0; off >>= 1) s += __shfl_xor(s, off, 32);
  float mean = s * (1.0f / EMB);
  float vs = 0.0f;
#pragma unroll
  for (int i = 0; i < 8; ++i) { float d = v[i] - mean; vs += d * d; }
#pragma unroll
  for (int off = 16; off > 0; off >>= 1) vs += __shfl_xor(vs, off, 32);
  float inv = rsqrtf(vs * (1.0f / EMB) + 1e-5f);
#pragma unroll
  for (int i = 0; i < 8; ++i) {
    int c = lane * 8 + i;
    float o = (v[i] - mean) * inv * g[c] + b[c];
    out [(size_t)row * EMB + c] = o;
    outh[(size_t)row * EMB + c] = (_Float16)o;
  }
}

// ---------------------------------------------------------------------------
// Fused masked attention, causally truncated: one wave per (b, h, q<=t).
// 2 keys/lane, shuffle softmax, probs in LDS, 1 output dim/lane (f16 out).
// ---------------------------------------------------------------------------
__global__ __launch_bounds__(256) void attention_kernel(
    const float* Q, const float* Kb, const float* Vb, _Float16* Oh, int t)
{
  __shared__ float probs[8][SEQ];
  int wave = threadIdx.x >> 5;
  int lane = threadIdx.x & 31;
  int qpLim = t + 1;
  int idx  = blockIdx.x * 8 + wave;
  if (idx >= BS * NH * qpLim) return;
  int b  = idx / (NH * qpLim);
  int rm = idx - b * NH * qpLim;
  int h  = rm / qpLim;
  int qp = rm - h * qpLim;

  const float* qv = Q  + ((size_t)(b * SEQ + qp) * EMB) + h * DH;
  const float* k0 = Kb + ((size_t)(b * SEQ + lane)      * EMB) + h * DH;
  const float* k1 = Kb + ((size_t)(b * SEQ + lane + 32) * EMB) + h * DH;
  float s0 = 0.0f, s1 = 0.0f;
#pragma unroll
  for (int d = 0; d < DH; ++d) {
    float qd = qv[d];
    s0 += qd * k0[d];
    s1 += qd * k1[d];
  }
  const float scale = 0.17677669529663687f;     // 1/sqrt(32)
  s0 *= scale; s1 *= scale;
  if (lane      > t) s0 = -1.0e9f;
  if (lane + 32 > t) s1 = -1.0e9f;
  float m = fmaxf(s0, s1);
#pragma unroll
  for (int off = 16; off > 0; off >>= 1) m = fmaxf(m, __shfl_xor(m, off, 32));
  float e0 = __expf(s0 - m), e1 = __expf(s1 - m);
  float sum = e0 + e1;
#pragma unroll
  for (int off = 16; off > 0; off >>= 1) sum += __shfl_xor(sum, off, 32);
  float inv = 1.0f / sum;
  probs[wave][lane]      = e0 * inv;
  probs[wave][lane + 32] = e1 * inv;
  __syncthreads();

  float acc = 0.0f;                             // lane == output dim d
  for (int kp = 0; kp <= t; ++kp)
    acc += probs[wave][kp] * Vb[((size_t)(b * SEQ + kp) * EMB) + h * DH + lane];
  Oh[((size_t)(b * SEQ + qp) * EMB) + h * DH + lane] = (_Float16)acc;
}

// ---------------------------------------------------------------------------
// Vocab softmax: writes d_out[b,t,:] (f32) and smbh[b,:] (f16, emb GEMM input)
// ---------------------------------------------------------------------------
__global__ __launch_bounds__(256) void softmax_head_kernel(
    const float* logits, float* outp, _Float16* smbh, int t)
{
  __shared__ float red[256];
  int b = blockIdx.x, tid = threadIdx.x;
  const float* lr = logits + (size_t)b * VOC;
  float mx = -3.4e38f;
  for (int i = tid; i < VOC; i += 256) mx = fmaxf(mx, lr[i]);
  red[tid] = mx; __syncthreads();
  for (int s = 128; s > 0; s >>= 1) { if (tid < s) red[tid] = fmaxf(red[tid], red[tid + s]); __syncthreads(); }
  mx = red[0]; __syncthreads();
  float sum = 0.0f;
  for (int i = tid; i < VOC; i += 256) sum += __expf(lr[i] - mx);
  red[tid] = sum; __syncthreads();
  for (int s = 128; s > 0; s >>= 1) { if (tid < s) red[tid] += red[tid + s]; __syncthreads(); }
  float inv = 1.0f / red[0];
  float*     o  = outp + (size_t)b * SEQ * VOC + (size_t)t * VOC;
  _Float16*  sb = smbh + (size_t)b * VOC;
  for (int i = tid; i < VOC; i += 256) {
    float p = __expf(lr[i] - mx) * inv;
    o[i]  = p;
    sb[i] = (_Float16)p;
  }
}

// ---------------------------------------------------------------------------
// Host orchestration
// ---------------------------------------------------------------------------
static void cvt(const float* s, _Float16* d, int n, hipStream_t st) {
  cvt_f16_kernel<<<(n + 255) / 256, 256, 0, st>>>(s, d, n);
}

extern "C" void kernel_launch(void* const* d_in, const int* in_sizes, int n_in,
                              void* d_out, int out_size, void* d_ws, size_t ws_size,
                              hipStream_t stream)
{
  (void)in_sizes; (void)n_in; (void)out_size; (void)ws_size;
  const float* noise   = (const float*)d_in[0];
  const float* seeds   = (const float*)d_in[1];
  const float* inemb_w = (const float*)d_in[2];
  const float* inemb_b = (const float*)d_in[3];
  const float* g_sa_w  = (const float*)d_in[4];
  const float* g_sa_b  = (const float*)d_in[5];
  const float* g_ca_w  = (const float*)d_in[6];
  const float* g_ca_b  = (const float*)d_in[7];
  const float* g_ff_w1 = (const float*)d_in[8];
  const float* g_ff_b1 = (const float*)d_in[9];
  const float* g_ff_w2 = (const float*)d_in[10];
  const float* g_ff_b2 = (const float*)d_in[11];
  const float* g_ln_g  = (const float*)d_in[12];
  const float* g_ln_b  = (const float*)d_in[13];
  const float* o_sa_w  = (const float*)d_in[14];
  const float* o_sa_b  = (const float*)d_in[15];
  const float* o_ff_w1 = (const float*)d_in[16];
  const float* o_ff_b1 = (const float*)d_in[17];
  const float* o_ff_w2 = (const float*)d_in[18];
  const float* o_ff_b2 = (const float*)d_in[19];
  const float* o_ln_g  = (const float*)d_in[20];
  const float* o_ln_b  = (const float*)d_in[21];
  const float* w2v_w   = (const float*)d_in[22];
  const float* soft_w  = (const float*)d_in[23];
  const float* soft_b  = (const float*)d_in[24];
  float* out = (float*)d_out;

  const int TOK = BS * SEQ;            // 1024 rows
  const int EE  = EMB * EMB;

  // ---- workspace carve-up: f32 region first (16B aligned throughout) ----
  float* wsf = (float*)d_ws;
  float* pe   = wsf;  wsf += SEQ * EMB;
  float* Y    = wsf;  wsf += TOK * EMB;
  float* x    = wsf;  wsf += TOK * EMB;
  float* q    = wsf;  wsf += TOK * EMB;
  float* kb   = wsf;  wsf += TOK * EMB;
  float* vb   = wsf;  wsf += TOK * EMB;
  float* pr   = wsf;  wsf += TOK * EMB;
  float* kc0  = wsf;  wsf += TOK * EMB;
  float* vc0  = wsf;  wsf += TOK * EMB;
  float* kc1  = wsf;  wsf += TOK * EMB;
  float* vc1  = wsf;  wsf += TOK * EMB;
  float* lg   = wsf;  wsf += BS * VOC;
  float* kc[2] = { kc0, kc1 };
  float* vc[2] = { vc0, vc1 };

  // ---- f16 region: activation mirrors + converted weights ----
  _Float16* wsh = (_Float16*)wsf;
  _Float16* noiseh = wsh;  wsh += TOK * EMB;
  _Float16* tmph   = wsh;  wsh += TOK * EMB;
  _Float16* Wmh    = wsh;  wsh += TOK * EMB;   // encoder memory (f16)
  _Float16* xh     = wsh;  wsh += TOK * EMB;
  _Float16* ath    = wsh;  wsh += TOK * EMB;
  _Float16* hidh   = wsh;  wsh += TOK * HID;
  _Float16* smbh   = wsh;  wsh += BS * VOC;
  _Float16* inembh = wsh;  wsh += 2 * EE;
  _Float16* gsawh  = wsh;  wsh += 8 * EE;
  _Float16* gcawh  = wsh;  wsh += 8 * EE;
  _Float16* gff1h  = wsh;  wsh += 2 * HID * EMB;
  _Float16* gff2h  = wsh;  wsh += 2 * EMB * HID;
  _Float16* osawh  = wsh;  wsh += 8 * EE;
  _Float16* off1h  = wsh;  wsh += 2 * HID * EMB;
  _Float16* off2h  = wsh;  wsh += 2 * EMB * HID;
  _Float16* softwh = wsh;  wsh += VOC * EMB;
  _Float16* w2vh   = wsh;  wsh += EMB * VOC;

  // ---- one-time conversions (weights are reused 64x; halves L2 traffic) ----
  cvt(noise,   noiseh, TOK * EMB,      stream);
  cvt(inemb_w, inembh, 2 * EE,         stream);
  cvt(g_sa_w,  gsawh,  8 * EE,         stream);
  cvt(g_ca_w,  gcawh,  8 * EE,         stream);
  cvt(g_ff_w1, gff1h,  2 * HID * EMB,  stream);
  cvt(g_ff_w2, gff2h,  2 * EMB * HID,  stream);
  cvt(o_sa_w,  osawh,  8 * EE,         stream);
  cvt(o_ff_w1, off1h,  2 * HID * EMB,  stream);
  cvt(o_ff_w2, off2h,  2 * EMB * HID,  stream);
  cvt(soft_w,  softwh, VOC * EMB,      stream);
  cvt(w2v_w,   w2vh,   EMB * VOC,      stream);

  pe_init_kernel<<<SEQ, EMB, 0, stream>>>(pe);
  zero_kernel<<<(TOK * EMB + 255) / 256, 256, 0, stream>>>(Y, TOK * EMB);

  // w = (noise @ inemb0.T + b0) @ inemb1.T + b1  (f16 mirrors only)
  gemm(noiseh, EMB, inembh,      inemb_b,       nullptr, tmph, EMB, TOK/16, 4, EMB, EMB, 0, stream);
  gemm(tmph,   EMB, inembh + EE, inemb_b + EMB, nullptr, Wmh,  EMB, TOK/16, 4, EMB, EMB, 0, stream);

  // cross-attention K/V are step-invariant: hoist out of the loop
  for (int i = 0; i < 2; ++i) {
    gemm(Wmh, EMB, gcawh + (size_t)(i * 4 + 1) * EE, g_ca_b + (i * 4 + 1) * EMB,
         kc[i], nullptr, EMB, TOK/16, 4, EMB, EMB, 0, stream);
    gemm(Wmh, EMB, gcawh + (size_t)(i * 4 + 2) * EE, g_ca_b + (i * 4 + 2) * EMB,
         vc[i], nullptr, EMB, TOK/16, 4, EMB, EMB, 0, stream);
  }

  for (int t = 0; t < SEQ; ++t) {
    const int sT    = (t >> 4) + 1;            // active 16-row s-tiles (causal)
    const int mT    = BS * sT;                 // active M-tiles per GEMM
    const int sRows = t + 1;                   // active s rows
    const int lnBlk = (BS * sRows + 7) / 8;
    const int atBlk = (BS * NH * sRows + 7) / 8;

    step_begin_kernel<<<(TOK * EMB) / 256, 256, 0, stream>>>(Y, seeds, pe, x, xh, t);

    // ---- generator blocks: SA + CA + FF ----
    for (int i = 0; i < 2; ++i) {
      const _Float16* Wsa = gsawh + (size_t)i * 4 * EE;
      const float*    Bsa = g_sa_b + i * 4 * EMB;
      gemm(xh, EMB, Wsa + 0 * EE, Bsa + 0 * EMB, q,  nullptr, EMB, mT, sT, EMB, EMB, 0, stream);
      gemm(xh, EMB, Wsa + 1 * EE, Bsa + 1 * EMB, kb, nullptr, EMB, mT, sT, EMB, EMB, 0, stream);
      gemm(xh, EMB, Wsa + 2 * EE, Bsa + 2 * EMB, vb, nullptr, EMB, mT, sT, EMB, EMB, 0, stream);
      attention_kernel<<<atBlk, 256, 0, stream>>>(q, kb, vb, ath, t);
      gemm(ath, EMB, Wsa + 3 * EE, Bsa + 3 * EMB, pr, nullptr, EMB, mT, sT, EMB, EMB, 0, stream);
      add_ln_kernel<<<lnBlk, 256, 0, stream>>>(pr, x,
          g_ln_g + (i * 3 + 0) * EMB, g_ln_b + (i * 3 + 0) * EMB, x, xh, sRows);

      const _Float16* Wca = gcawh + (size_t)i * 4 * EE;
      const float*    Bca = g_ca_b + i * 4 * EMB;
      gemm(xh, EMB, Wca + 0 * EE, Bca + 0 * EMB, q, nullptr, EMB, mT, sT, EMB, EMB, 0, stream);
      attention_kernel<<<atBlk, 256, 0, stream>>>(q, kc[i], vc[i], ath, t);
      gemm(ath, EMB, Wca + 3 * EE, Bca + 3 * EMB, pr, nullptr, EMB, mT, sT, EMB, EMB, 0, stream);
      add_ln_kernel<<<lnBlk, 256, 0, stream>>>(pr, x,
          g_ln_g + (i * 3 + 1) * EMB, g_ln_b + (i * 3 + 1) * EMB, x, xh, sRows);

      gemm(xh,   EMB, gff1h + (size_t)i * HID * EMB, g_ff_b1 + i * HID,
           nullptr, hidh, HID, mT, sT, HID, EMB, 1, stream);          // ReLU fused
      gemm(hidh, HID, gff2h + (size_t)i * EMB * HID, g_ff_b2 + i * EMB,
           pr, nullptr, EMB, mT, sT, EMB, HID, 0, stream);
      add_ln_kernel<<<lnBlk, 256, 0, stream>>>(pr, x,
          g_ln_g + (i * 3 + 2) * EMB, g_ln_b + (i * 3 + 2) * EMB, x, xh, sRows);
    }

    // ---- output blocks: SA + FF ----
    for (int i = 0; i < 2; ++i) {
      const _Float16* Wsa = osawh + (size_t)i * 4 * EE;
      const float*    Bsa = o_sa_b + i * 4 * EMB;
      gemm(xh, EMB, Wsa + 0 * EE, Bsa + 0 * EMB, q,  nullptr, EMB, mT, sT, EMB, EMB, 0, stream);
      gemm(xh, EMB, Wsa + 1 * EE, Bsa + 1 * EMB, kb, nullptr, EMB, mT, sT, EMB, EMB, 0, stream);
      gemm(xh, EMB, Wsa + 2 * EE, Bsa + 2 * EMB, vb, nullptr, EMB, mT, sT, EMB, EMB, 0, stream);
      attention_kernel<<<atBlk, 256, 0, stream>>>(q, kb, vb, ath, t);
      gemm(ath, EMB, Wsa + 3 * EE, Bsa + 3 * EMB, pr, nullptr, EMB, mT, sT, EMB, EMB, 0, stream);
      add_ln_kernel<<<lnBlk, 256, 0, stream>>>(pr, x,
          o_ln_g + (i * 2 + 0) * EMB, o_ln_b + (i * 2 + 0) * EMB, x, xh, sRows);

      gemm(xh,   EMB, off1h + (size_t)i * HID * EMB, o_ff_b1 + i * HID,
           nullptr, hidh, HID, mT, sT, HID, EMB, 1, stream);
      gemm(hidh, HID, off2h + (size_t)i * EMB * HID, o_ff_b2 + i * EMB,
           pr, nullptr, EMB, mT, sT, EMB, HID, 0, stream);
      add_ln_kernel<<<lnBlk, 256, 0, stream>>>(pr, x,
          o_ln_g + (i * 2 + 1) * EMB, o_ln_b + (i * 2 + 1) * EMB, x, xh, sRows);
    }

    // ---- head: logits (row extraction via lda), vocab softmax, feedback ----
    gemm(xh + (size_t)t * EMB, SEQ * EMB, softwh, soft_b, lg, nullptr, VOC,
         1, 1, VOC, EMB, 0, stream);
    softmax_head_kernel<<<BS, 256, 0, stream>>>(lg, out, smbh, t);
    gemm(smbh, VOC, w2vh, nullptr, Y + (size_t)t * EMB, nullptr, SEQ * EMB,
         1, 1, EMB, VOC, 0, stream);                                  // Y[:,t] = emb
  }
}
